// RelationalGINEncoder_81209241633067
// MI455X (gfx1250) — compile-verified, matrix operand order checked by reading, not run on hardware
//
#include <hip/hip_runtime.h>
#include <hip/hip_bf16.h>
#include <stdint.h>

#define N_NODES 100000
#define N_EDGES 640000
#define N_REL   64
#define DIM     128
#define DIM2    256

typedef __attribute__((ext_vector_type(16))) _Float16 v16h;
typedef __attribute__((ext_vector_type(8)))  _Float16 v8h;
typedef __attribute__((ext_vector_type(8)))  float    v8f;

// ---------------------------------------------------------------------------
// WMMA f16 fragment layout (V_WMMA_F32_16X16X32_F16, gfx1250):
// lane = (r, half): r = lane&15 (A: row M / B: col N), half = lane>>4.
// element e (0..15): k_local = 8*half + e + ((e>=8)?8:0)
//   -> two contiguous 8-element runs at k0+8*half and k0+8*half+16.
// hi/lo f16 split (Markidis) for near-f32 accuracy, f32 accumulate.
// ---------------------------------------------------------------------------
__device__ __forceinline__ void load_frag_pair(const float* __restrict__ row,
                                               int k0, int half,
                                               v16h& hi, v16h& lo) {
#pragma unroll
  for (int e = 0; e < 16; ++e) {
    int k = k0 + 8 * half + e + ((e & 8) ? 8 : 0);
    float f = row[k];
    _Float16 fh = (_Float16)f;
    hi[e] = fh;
    lo[e] = (_Float16)(f - (float)fh);
  }
}

// fused (1+eps)*x + aggr version (builds GEMM input without materializing it)
__device__ __forceinline__ void load_frag_pair_comb(const float* __restrict__ xrow,
                                                    const float* __restrict__ grow,
                                                    float c, int k0, int half,
                                                    v16h& hi, v16h& lo) {
#pragma unroll
  for (int e = 0; e < 16; ++e) {
    int k = k0 + 8 * half + e + ((e & 8) ? 8 : 0);
    float f = c * xrow[k] + grow[k];
    _Float16 fh = (_Float16)f;
    hi[e] = fh;
    lo[e] = (_Float16)(f - (float)fh);
  }
}

// load a fragment from pre-split f16 storage: two contiguous v8h runs
__device__ __forceinline__ v16h load_frag_f16(const _Float16* __restrict__ base,
                                              int k0, int half) {
  const _Float16* p = base + k0 + 8 * half;
  v8h a = *(const v8h*)p;
  v8h b = *(const v8h*)(p + 16);
  return __builtin_shufflevector(a, b, 0, 1, 2, 3, 4, 5, 6, 7,
                                 8, 9, 10, 11, 12, 13, 14, 15);
}

__device__ __forceinline__ v8f wmma3(v8f acc, v16h ahi, v16h alo, v16h bhi, v16h blo) {
  acc = __builtin_amdgcn_wmma_f32_16x16x32_f16(false, ahi, false, bhi, (short)0, acc, false, false);
  acc = __builtin_amdgcn_wmma_f32_16x16x32_f16(false, ahi, false, blo, (short)0, acc, false, false);
  acc = __builtin_amdgcn_wmma_f32_16x16x32_f16(false, alo, false, bhi, (short)0, acc, false, false);
  return acc;
}

// ---------------------------------------------------------------------------
// Weight prepack: W (Nout x K) row-major f32 -> per-(nt,ks) fragments.
// Fragment f = nt*nks+ks occupies 1024 f16: [lane][16] hi at +0, lo at +512.
// Done once per layer; GEMM loops then do pure 32B vector loads.
// ---------------------------------------------------------------------------
__global__ void prepack_kernel(const float* __restrict__ W,
                               _Float16* __restrict__ out,
                               int ntiles, int nks) {
  int t = blockIdx.x * blockDim.x + threadIdx.x;
  int total = ntiles * nks * 32;
  if (t >= total) return;
  int lane = t & 31;
  int f = t >> 5;
  int ks = f % nks;
  int nt = f / nks;
  int r = lane & 15, half = lane >> 4;
  int K = nks * 32;
  const float* row = W + (size_t)(nt * 16 + r) * K + ks * 32;
  _Float16* hi = out + (size_t)f * 1024 + lane * 16;
  _Float16* lo = hi + 512;
#pragma unroll
  for (int e = 0; e < 16; ++e) {
    int k = 8 * half + e + ((e & 8) ? 8 : 0);
    float v = row[k];
    _Float16 h = (_Float16)v;
    hi[e] = h;
    lo[e] = (_Float16)(v - (float)h);
  }
}

__device__ __forceinline__ void load_packed(const _Float16* __restrict__ pack,
                                            int f, int lane, v16h& hi, v16h& lo) {
  const _Float16* fb = pack + (size_t)f * 1024 + lane * 16;
  hi = *(const v16h*)fb;
  lo = *(const v16h*)(fb + 512);
}

// ---------------------------------------------------------------------------
// Relation MLP (tiny): one block per relation row.
// ---------------------------------------------------------------------------
__global__ void relmlp_kernel(const float* __restrict__ rel,
                              const float* __restrict__ w0, const float* __restrict__ b0,
                              const float* __restrict__ w1, const float* __restrict__ b1,
                              const float* __restrict__ w2, const float* __restrict__ b2,
                              const float* __restrict__ w3, const float* __restrict__ b3,
                              float* __restrict__ gamma, float* __restrict__ beta) {
  __shared__ float ha[DIM2];
  __shared__ float hb[DIM2];
  const int r = blockIdx.x;
  const int j = threadIdx.x;
  if (j < DIM) ha[j] = rel[r * DIM + j];
  __syncthreads();
  {
    float s = b0[j];
    const float* wr = w0 + j * DIM;
#pragma unroll 4
    for (int k = 0; k < DIM; ++k) s += ha[k] * wr[k];
    hb[j] = fmaxf(s, 0.f);
  }
  __syncthreads();
  {
    float s = b1[j];
    const float* wr = w1 + j * DIM2;
#pragma unroll 4
    for (int k = 0; k < DIM2; ++k) s += hb[k] * wr[k];
    ha[j] = fmaxf(s, 0.f);
  }
  __syncthreads();
  {
    float s = b2[j];
    const float* wr = w2 + j * DIM2;
#pragma unroll 4
    for (int k = 0; k < DIM2; ++k) s += ha[k] * wr[k];
    hb[j] = fmaxf(s, 0.f);
  }
  __syncthreads();
  {
    float s = b3[j];
    const float* wr = w3 + j * DIM2;
#pragma unroll 4
    for (int k = 0; k < DIM2; ++k) s += hb[k] * wr[k];
    if (j < DIM) gamma[r * DIM + j] = s;
    else         beta[r * DIM + (j - DIM)] = s;
  }
}

// ---------------------------------------------------------------------------
// xw = x @ W^T + b  (M=100000, N=128, K=128). One wave per 16-row tile.
// B side: prepacked fragments (pure global_load_b128, no conversion VALU).
// ---------------------------------------------------------------------------
__global__ void xw_kernel(const float* __restrict__ x,
                          const _Float16* __restrict__ wpack,  // 8 nt x 4 ks frags
                          const float* __restrict__ bias,
                          float* __restrict__ out, int nrows) {
  const int wave = (int)((blockIdx.x * blockDim.x + threadIdx.x) >> 5);
  const int lane = threadIdx.x & 31;
  const int m0 = wave * 16;
  if (m0 >= nrows) return;  // wave-uniform -> EXEC stays all ones
  const int r = lane & 15, half = lane >> 4;

  const float* arow = x + (size_t)(m0 + r) * DIM;
  v16h ahi[4], alo[4];
#pragma unroll
  for (int ks = 0; ks < 4; ++ks) load_frag_pair(arow, ks * 32, half, ahi[ks], alo[ks]);

  for (int nt = 0; nt < 8; ++nt) {
    float bv = bias[nt * 16 + r];
    v8f acc;
#pragma unroll
    for (int i = 0; i < 8; ++i) acc[i] = bv;
#pragma unroll
    for (int ks = 0; ks < 4; ++ks) {
      v16h bhi, blo;
      load_packed(wpack, nt * 4 + ks, lane, bhi, blo);
      acc = wmma3(acc, ahi[ks], alo[ks], bhi, blo);
    }
    float* obase = out + (size_t)m0 * DIM + nt * 16;
#pragma unroll
    for (int i = 0; i < 8; ++i) {
      int m = i + 8 * half;
      obase[(size_t)m * DIM + r] = acc[i];
    }
  }
}

__global__ void zero_kernel(float* __restrict__ p, long long n) {
  long long i = (long long)blockIdx.x * blockDim.x + threadIdx.x;
  if (i < n) p[i] = 0.f;
}

// ---------------------------------------------------------------------------
// Edge phase: m = gamma[t]*xw[src] + beta[t]; aggr[dst] += m.
// One wave/edge, float4/lane, hardware f32 atomics (aggr+xw are L2-resident
// on the 192MB L2 -> random gather/scatter never touches HBM).
// ---------------------------------------------------------------------------
__global__ void edge_kernel(const int* __restrict__ ei, const int* __restrict__ et,
                            const float* __restrict__ xw,
                            const float* __restrict__ gamma,
                            const float* __restrict__ beta,
                            float* __restrict__ aggr, int nedges) {
  const int gw = (int)((blockIdx.x * blockDim.x + threadIdx.x) >> 5);
  const int lane = threadIdx.x & 31;
  if (gw >= nedges) return;
  const int src = ei[gw];
  const int dst = ei[nedges + gw];
  const int t   = et[gw];
  const float4 xv = ((const float4*)(xw + (size_t)src * DIM))[lane];
  const float4 gv = ((const float4*)(gamma + (size_t)t * DIM))[lane];
  const float4 bv = ((const float4*)(beta + (size_t)t * DIM))[lane];
  float m0 = gv.x * xv.x + bv.x;
  float m1 = gv.y * xv.y + bv.y;
  float m2 = gv.z * xv.z + bv.z;
  float m3 = gv.w * xv.w + bv.w;
  float* ad = aggr + (size_t)dst * DIM + lane * 4;
  __hip_atomic_fetch_add(ad + 0, m0, __ATOMIC_RELAXED, __HIP_MEMORY_SCOPE_AGENT);
  __hip_atomic_fetch_add(ad + 1, m1, __ATOMIC_RELAXED, __HIP_MEMORY_SCOPE_AGENT);
  __hip_atomic_fetch_add(ad + 2, m2, __ATOMIC_RELAXED, __HIP_MEMORY_SCOPE_AGENT);
  __hip_atomic_fetch_add(ad + 3, m3, __ATOMIC_RELAXED, __HIP_MEMORY_SCOPE_AGENT);
}

// ---------------------------------------------------------------------------
// Fused GIN update + 2-layer MLP, one wave per 16-row tile, 4 waves/block.
// LDS (192KB of WGP's 320KB):
//   [0,128K):    mlp_w0 fragments, block-staged via GLOBAL_LOAD_ASYNC_TO_LDS_B128
//                (ASYNCcnt) and shared by all 4 waves.
//   [128K,192K): 4 wave-private h tiles, PRE-SPLIT f16 hi/lo (16x256 each),
//                so GEMM2 A-fragments are pure 16B ds loads, no reconversion.
// ---------------------------------------------------------------------------
#define W0PACK_BYTES (16 * 4 * 2 * 1024)  // 16 nt * 4 ks * (hi+lo) * 1KB = 128KB
#define HTILE_HALF   (16 * DIM2)          // 4096 f16 per hi (or lo) plane

__global__ void gin_mlp_kernel(const float* __restrict__ x,
                               const float* __restrict__ aggr,
                               const float* __restrict__ epsp,
                               const _Float16* __restrict__ w0pack,  // 16x4 frags
                               const float* __restrict__ b0,
                               const _Float16* __restrict__ w1pack,  // 8x8 frags
                               const float* __restrict__ b1,
                               float* __restrict__ xnext, int nrows) {
  extern __shared__ float smem[];
  const int wavein = threadIdx.x >> 5;
  const int wave = (int)(blockIdx.x * (blockDim.x >> 5) + wavein);
  const int lane = threadIdx.x & 31;
  const int r = lane & 15, half = lane >> 4;

  // ---- stage w0 fragments into LDS (async, block-wide) ----
  {
    const char* gsrc = (const char*)w0pack;
    uint32_t lbase = (uint32_t)(uintptr_t)smem;  // LDS byte offset of dyn-shared
    const int nloads = W0PACK_BYTES / 16;        // 8192 x b128
#pragma unroll
    for (int it = 0; it < nloads / 128; ++it) {  // 64 iters, 128 threads
      int idx = it * 128 + (int)threadIdx.x;
      uint32_t loff = lbase + (uint32_t)idx * 16u;
      uint64_t ga = (uint64_t)(uintptr_t)(gsrc + (size_t)idx * 16);
      asm volatile("global_load_async_to_lds_b128 %0, %1, off"
                   :: "v"(loff), "v"(ga) : "memory");
    }
    asm volatile("s_wait_asynccnt 0" ::: "memory");
  }
  __syncthreads();

  const int m0 = wave * 16;
  if (m0 >= nrows) return;  // after the only barrier; wave-uniform

  const _Float16* w0lds = (const _Float16*)smem;
  _Float16* hhi = (_Float16*)((char*)smem + W0PACK_BYTES) +
                  (size_t)wavein * (2 * HTILE_HALF);
  _Float16* hlo = hhi + HTILE_HALF;
  const float onepe = 1.0f + epsp[0];

  // ---- GEMM1: h = relu(((1+eps)x + aggr) @ w0^T + b0), K=128, N=256 ----
  const float* xrow = x    + (size_t)(m0 + r) * DIM;
  const float* grow = aggr + (size_t)(m0 + r) * DIM;
  v16h ahi[4], alo[4];
#pragma unroll
  for (int ks = 0; ks < 4; ++ks)
    load_frag_pair_comb(xrow, grow, onepe, ks * 32, half, ahi[ks], alo[ks]);

  for (int nt = 0; nt < 16; ++nt) {
    float bv = b0[nt * 16 + r];
    v8f acc;
#pragma unroll
    for (int i = 0; i < 8; ++i) acc[i] = bv;
#pragma unroll
    for (int ks = 0; ks < 4; ++ks) {
      v16h bhi, blo;
      load_packed(w0lds, nt * 4 + ks, lane, bhi, blo);  // ds_load from LDS
      acc = wmma3(acc, ahi[ks], alo[ks], bhi, blo);
    }
    // relu + single hi/lo split at store time; D tile elem (m=i+8*half, n)
#pragma unroll
    for (int i = 0; i < 8; ++i) {
      float v = fmaxf(acc[i], 0.f);
      _Float16 vh = (_Float16)v;
      int idx = (i + 8 * half) * DIM2 + nt * 16 + r;
      hhi[idx] = vh;
      hlo[idx] = (_Float16)(v - (float)vh);
    }
  }
  // wave-private h region; DS ops are in-order within a wave -> no barrier.

  // ---- GEMM2: xnext = h @ w1^T + b1, K=256, N=128 ----
  // A-fragments: pure 16B ds loads from pre-split hi/lo planes.
  const _Float16* hhr = hhi + r * DIM2;
  const _Float16* hlr = hlo + r * DIM2;
  for (int nt = 0; nt < 8; ++nt) {
    float bv = b1[nt * 16 + r];
    v8f acc;
#pragma unroll
    for (int i = 0; i < 8; ++i) acc[i] = bv;
#pragma unroll
    for (int ks = 0; ks < 8; ++ks) {
      v16h a2hi = load_frag_f16(hhr, ks * 32, half);
      v16h a2lo = load_frag_f16(hlr, ks * 32, half);
      v16h bhi, blo;
      load_packed(w1pack, nt * 8 + ks, lane, bhi, blo);  // global, L2-resident
      acc = wmma3(acc, a2hi, a2lo, bhi, blo);
    }
    float* obase = xnext + (size_t)m0 * DIM + nt * 16;
#pragma unroll
    for (int i = 0; i < 8; ++i) {
      int m = i + 8 * half;
      obase[(size_t)m * DIM + r] = acc[i];
    }
  }
}

// ---------------------------------------------------------------------------
extern "C" void kernel_launch(void* const* d_in, const int* in_sizes, int n_in,
                              void* d_out, int out_size, void* d_ws, size_t ws_size,
                              hipStream_t stream) {
  const int*   ei     = (const int*)d_in[0];    // (2, E)
  const int*   et     = (const int*)d_in[1];    // (E,)
  const float* embed  = (const float*)d_in[2];  // (N, D)
  const float* rel    = (const float*)d_in[3];  // (L, R, D)
  const float* rmw0   = (const float*)d_in[4];
  const float* rmb0   = (const float*)d_in[5];
  const float* rmw1   = (const float*)d_in[6];
  const float* rmb1   = (const float*)d_in[7];
  const float* rmw2   = (const float*)d_in[8];
  const float* rmb2   = (const float*)d_in[9];
  const float* rmw3   = (const float*)d_in[10];
  const float* rmb3   = (const float*)d_in[11];
  const float* W_w    = (const float*)d_in[12]; // (L, D, D)
  const float* W_b    = (const float*)d_in[13];
  const float* mlp_w0 = (const float*)d_in[14]; // (L, 2D, D)
  const float* mlp_b0 = (const float*)d_in[15];
  const float* mlp_w1 = (const float*)d_in[16]; // (L, D, 2D)
  const float* mlp_b1 = (const float*)d_in[17];
  const float* eps    = (const float*)d_in[18]; // (L,)

  const long long ND = (long long)N_NODES * DIM;
  float* ws    = (float*)d_ws;
  float* xmid  = ws;               // x after layer 0
  float* xwbuf = ws + ND;          // xw
  float* aggr  = ws + 2 * ND;      // segment sum
  float* gamma = ws + 3 * ND;      // (R, D)
  float* beta  = gamma + (long long)N_REL * DIM;
  _Float16* wpackW = (_Float16*)(beta + (long long)N_REL * DIM); // 8*4  frags
  _Float16* wpack0 = wpackW + 32 * 1024;                         // 16*4 frags
  _Float16* wpack1 = wpack0 + 64 * 1024;                         // 8*8  frags

  const int mtiles   = N_NODES / 16;                   // 6250 exact
  const int xw_blks  = (mtiles * 32 + 255) / 256;
  const int mlp_blks = (mtiles + 3) / 4;
  const int z_blks   = (int)((ND + 255) / 256);
  const int e_blks   = (N_EDGES * 32) / 256;
  const size_t mlp_lds = W0PACK_BYTES + 4 * 2 * HTILE_HALF * sizeof(_Float16); // 192KB

  const float* xcur = embed;
  for (int l = 0; l < 2; ++l) {
    relmlp_kernel<<<N_REL, 256, 0, stream>>>(
        rel + (long long)l * N_REL * DIM,
        rmw0 + (long long)l * DIM2 * DIM,  rmb0 + l * DIM2,
        rmw1 + (long long)l * DIM2 * DIM2, rmb1 + l * DIM2,
        rmw2 + (long long)l * DIM2 * DIM2, rmb2 + l * DIM2,
        rmw3 + (long long)l * DIM2 * DIM2, rmb3 + l * DIM2,
        gamma, beta);

    // prepack this layer's GEMM weights into WMMA-ready f16 hi/lo fragments
    prepack_kernel<<<(8 * 4 * 32 + 255) / 256, 256, 0, stream>>>(
        W_w + (long long)l * DIM * DIM, wpackW, 8, 4);
    prepack_kernel<<<(16 * 4 * 32 + 255) / 256, 256, 0, stream>>>(
        mlp_w0 + (long long)l * DIM2 * DIM, wpack0, 16, 4);
    prepack_kernel<<<(8 * 8 * 32 + 255) / 256, 256, 0, stream>>>(
        mlp_w1 + (long long)l * DIM * DIM2, wpack1, 8, 8);

    xw_kernel<<<xw_blks, 256, 0, stream>>>(
        xcur, wpackW, W_b + l * DIM, xwbuf, N_NODES);

    zero_kernel<<<z_blks, 256, 0, stream>>>(aggr, ND);

    edge_kernel<<<e_blks, 256, 0, stream>>>(ei, et, xwbuf, gamma, beta, aggr, N_EDGES);

    float* xnext = (l == 0) ? xmid : (float*)d_out;
    gin_mlp_kernel<<<mlp_blks, 128, mlp_lds, stream>>>(
        xcur, aggr, eps + l,
        wpack0, mlp_b0 + l * DIM2,
        wpack1, mlp_b1 + l * DIM,
        xnext, N_NODES);
    xcur = xnext;
  }
}
